// SparseAttention_44985487458308
// MI455X (gfx1250) — compile-verified, matrix operand order checked by reading, not run on hardware
//
#include <hip/hip_runtime.h>
#include <hip/hip_bf16.h>

#define S_LEN 2048
#define EMB   1024
#define NH    16
#define HD    64
#define BATCH 2
#define BHCNT (BATCH*NH)
#define TOPK  419430
#define NBINS 4096

typedef __attribute__((ext_vector_type(16))) __bf16 v16bf;
typedef __attribute__((ext_vector_type(8)))  float  v8f;
typedef __attribute__((ext_vector_type(4)))  float  f4;

union Frag16 { v16bf v; uint4 q[2]; unsigned short s[16]; };

__device__ __forceinline__ unsigned short f2bf(float f) {
  unsigned int x = __float_as_uint(f);
  unsigned int r = x + 0x7FFFu + ((x >> 16) & 1u);   // round-to-nearest-even
  return (unsigned short)(r >> 16);
}

// NT fragment load for 16-bit WMMA A/B operands.
// Lane holds row (row + (lane&15)); K elements {k0+half*8+0..7, k0+16+half*8+0..7}
__device__ __forceinline__ v16bf ld_frag(const unsigned short* base, int ld,
                                         int row, int k0, int lane) {
  int m = lane & 15, half = lane >> 4;
  const unsigned short* p = base + (size_t)(row + m) * ld + k0 + half * 8;
  Frag16 f;
  f.q[0] = *(const uint4*)(p);
  f.q[1] = *(const uint4*)(p + 16);
  return f.v;
}

__device__ __forceinline__ v8f wmma_bf16(v16bf a, v16bf b, v8f c) {
  return __builtin_amdgcn_wmma_f32_16x16x32_bf16(false, a, false, b, (short)0, c, false, false);
}

__global__ void zero_i32(int* p, int n) {
  int i = blockIdx.x * blockDim.x + threadIdx.x;
  if (i < n) p[i] = 0;
}

__global__ void cvt_bf16(const float* __restrict__ in, unsigned short* __restrict__ out, int n) {
  int i = blockIdx.x * blockDim.x + threadIdx.x;
  if (i < n) out[i] = f2bf(in[i]);
}

__device__ __forceinline__ void load_ab(const unsigned short* __restrict__ A,
                                        const unsigned short* __restrict__ W,
                                        int mbase, int nbase, int k0, int lane,
                                        v16bf a[2], v16bf b[4]) {
  a[0] = ld_frag(A, EMB, mbase,      k0, lane);
  a[1] = ld_frag(A, EMB, mbase + 16, k0, lane);
#pragma unroll
  for (int t = 0; t < 4; ++t) b[t] = ld_frag(W, EMB, nbase + t * 16, k0, lane);
}

__device__ __forceinline__ void mma_set(v8f acc[8], const v16bf a[2], const v16bf b[4]) {
#pragma unroll
  for (int t = 0; t < 4; ++t) {
    acc[t]     = wmma_bf16(a[0], b[t], acc[t]);
    acc[4 + t] = wmma_bf16(a[1], b[t], acc[4 + t]);
  }
}

// C[M,N] = A[M,K] * W[N,K]^T + bias.  M=B*S, N=K=EMB.
// Wave tile 32x64; 2-deep ping-pong over K: each fragment set is consumed then
// reloaded in place, so loads for k+2 steps overlap WMMAs with no register copies.
// transposedV==0 : store bf16 as [M, N]  (Q / K projections)
// transposedV==1 : store bf16 as Vt [b, h, d, S]  (V projection, pre-transposed for P*V)
__global__ __launch_bounds__(128) void gemm_qkv(const unsigned short* __restrict__ A,
                                                const unsigned short* __restrict__ W,
                                                const float* __restrict__ bias,
                                                unsigned short* __restrict__ out,
                                                int transposedV) {
  const int N = EMB;
  int lane = threadIdx.x & 31, wave = threadIdx.x >> 5;
  int mbase = blockIdx.x * 32;
  int nbase = (blockIdx.y * 4 + wave) * 64;
  v8f zero = {0.f,0.f,0.f,0.f,0.f,0.f,0.f,0.f};
  v8f acc[8];
#pragma unroll
  for (int i = 0; i < 8; ++i) acc[i] = zero;

  v16bf a0[2], b0[4], a1[2], b1[4];
  load_ab(A, W, mbase, nbase, 0,  lane, a0, b0);
  load_ab(A, W, mbase, nbase, 32, lane, a1, b1);

  const int KS = EMB / 32;                 // 32 k-steps
  for (int ks = 0; ks < KS - 2; ks += 2) {
    mma_set(acc, a0, b0);
    load_ab(A, W, mbase, nbase, (ks + 2) * 32, lane, a0, b0);
    mma_set(acc, a1, b1);
    load_ab(A, W, mbase, nbase, (ks + 3) * 32, lane, a1, b1);
  }
  mma_set(acc, a0, b0);
  mma_set(acc, a1, b1);

  int m = lane & 15, half = lane >> 4;
#pragma unroll
  for (int r = 0; r < 2; ++r) {
#pragma unroll
    for (int t = 0; t < 4; ++t) {
#pragma unroll
      for (int j = 0; j < 8; ++j) {
        int row = mbase + r * 16 + j + 8 * half;
        int col = nbase + t * 16 + m;
        unsigned short bv = f2bf(acc[r * 4 + t][j] + bias[col]);
        if (!transposedV) {
          out[(size_t)row * N + col] = bv;
        } else {
          int b = row / S_LEN, s = row - b * S_LEN;
          int h = col >> 6, dd = col & 63;
          out[(((size_t)(b * NH + h)) * HD + dd) * S_LEN + s] = bv;
        }
      }
    }
  }
}

// scores[bh,q,k] = (Q_h . K_h^T)/8 ; also builds per-(b,h) histogram for top-k threshold.
// Score stores are non-temporal: the 536MB map is streamed and must not evict
// the L2-resident bf16 Q/K/V working set.
__global__ __launch_bounds__(128) void scores_hist(const unsigned short* __restrict__ Q,
                                                   const unsigned short* __restrict__ Kp,
                                                   float* __restrict__ sc,
                                                   int* __restrict__ hist) {
  __shared__ int lh[NBINS];
  for (int i = threadIdx.x; i < NBINS; i += 128) lh[i] = 0;
  __syncthreads();

  int lane = threadIdx.x & 31, wave = threadIdx.x >> 5;
  int bh = blockIdx.z, b = bh >> 4, h = bh & 15;
  int qbase = blockIdx.x * 16;
  int nbase = blockIdx.y * 256 + wave * 64;
  const unsigned short* Qb = Q  + (size_t)b * S_LEN * EMB + h * HD;
  const unsigned short* Kb = Kp + (size_t)b * S_LEN * EMB + h * HD;

  v8f zero = {0.f,0.f,0.f,0.f,0.f,0.f,0.f,0.f};
  v8f acc[4] = {zero, zero, zero, zero};
#pragma unroll
  for (int k0 = 0; k0 < HD; k0 += 32) {
    v16bf a = ld_frag(Qb, EMB, qbase, k0, lane);
#pragma unroll
    for (int t = 0; t < 4; ++t) {
      v16bf bb = ld_frag(Kb, EMB, nbase + t * 16, k0, lane);
      acc[t] = wmma_bf16(a, bb, acc[t]);
    }
  }

  float* base = sc + (size_t)bh * S_LEN * S_LEN;
  int m = lane & 15, half = lane >> 4;
#pragma unroll
  for (int t = 0; t < 4; ++t) {
#pragma unroll
    for (int j = 0; j < 8; ++j) {
      int q  = qbase + j + 8 * half;
      int kc = nbase + t * 16 + m;
      float v = acc[t][j] * 0.125f;            // 1/sqrt(64), TEMPERATURE=1
      __builtin_nontemporal_store(v, &base[(size_t)q * S_LEN + kc]);
      int bin = (int)((v + 8.0f) * 256.0f);
      bin = bin < 0 ? 0 : (bin > NBINS - 1 ? NBINS - 1 : bin);
      atomicAdd(&lh[bin], 1);
    }
  }
  __syncthreads();
  for (int i = threadIdx.x; i < NBINS; i += 128)
    if (lh[i]) atomicAdd(&hist[bh * NBINS + i], lh[i]);
}

__global__ void find_thresh(const int* __restrict__ hist, float* __restrict__ thr) {
  int bh = blockIdx.x;
  if (threadIdx.x == 0) {
    const int* hb = hist + bh * NBINS;
    long cum = 0;
    float tv = -8.0f;
    for (int i = NBINS - 1; i >= 0; --i) {
      cum += hb[i];
      if (cum >= TOPK) { tv = (float)i * (1.0f / 256.0f) - 8.0f; break; }
    }
    thr[bh] = tv;
  }
}

// ---- attn_out tile helpers (ping-pong pipelined) ----
#define LOAD_SCORES(SV, KT) {                                                  \
  float* _p = srow + (KT) * 32 + half * 8;                                     \
  *(f4*)&(SV)[0]  = __builtin_nontemporal_load((const f4*)(_p));               \
  *(f4*)&(SV)[4]  = __builtin_nontemporal_load((const f4*)(_p + 4));           \
  *(f4*)&(SV)[8]  = __builtin_nontemporal_load((const f4*)(_p + 16));          \
  *(f4*)&(SV)[12] = __builtin_nontemporal_load((const f4*)(_p + 20)); }

#define LOAD_VFRAG(BB, KT) {                                                   \
  _Pragma("unroll")                                                            \
  for (int _t = 0; _t < 4; ++_t)                                               \
    (BB)[_t] = ld_frag(Vb, S_LEN, _t * 16, (KT) * 32, lane); }

#define PROC_TILE(SV, BB, KT) {                                                \
  float ms[16], mk[16];                                                        \
  float tmax = -3.0e38f;                                                       \
  _Pragma("unroll")                                                            \
  for (int i = 0; i < 16; ++i) {                                               \
    bool keep = (SV)[i] >= th;                                                 \
    mk[i] = keep ? 1.0f : 0.0f;                                                \
    ms[i] = keep ? (SV)[i] : -1.0e9f;                                          \
    tmax = fmaxf(tmax, ms[i]);                                                 \
  }                                                                            \
  float* _p = srow + (KT) * 32 + half * 8;                                     \
  __builtin_nontemporal_store(*(const f4*)&mk[0],  (f4*)(_p));                 \
  __builtin_nontemporal_store(*(const f4*)&mk[4],  (f4*)(_p + 4));             \
  __builtin_nontemporal_store(*(const f4*)&mk[8],  (f4*)(_p + 16));            \
  __builtin_nontemporal_store(*(const f4*)&mk[12], (f4*)(_p + 20));            \
  float cross = fmaxf(tmax, __shfl_xor(tmax, 16));                             \
  if (half == 0) {                                                             \
    float old = rowmax[wave][m];                                               \
    float nm0 = fmaxf(old, cross);                                             \
    rowmax[wave][m]   = nm0;                                                   \
    float scl         = __expf(old - nm0);                                     \
    rowscale[wave][m] = scl;                                                   \
    rowsum[wave][m]  *= scl;                                                   \
  }                                                                            \
  __syncthreads();                                                             \
  float nm = rowmax[wave][m];                                                  \
  float psum = 0.0f;                                                           \
  Frag16 pf;                                                                   \
  _Pragma("unroll")                                                            \
  for (int i = 0; i < 16; ++i) {                                               \
    float p = __expf(ms[i] - nm);                                              \
    psum += p;                                                                 \
    pf.s[i] = f2bf(p);                                                         \
  }                                                                            \
  float tot = psum + __shfl_xor(psum, 16);                                     \
  float scl8[8];                                                               \
  _Pragma("unroll")                                                            \
  for (int j = 0; j < 8; ++j) scl8[j] = rowscale[wave][j + 8 * half];          \
  _Pragma("unroll")                                                            \
  for (int t = 0; t < 4; ++t)                                                  \
    _Pragma("unroll")                                                          \
    for (int j = 0; j < 8; ++j) acc[t][j] *= scl8[j];                          \
  if (half == 0) rowsum[wave][m] += tot;                                       \
  _Pragma("unroll")                                                            \
  for (int t = 0; t < 4; ++t) acc[t] = wmma_bf16(pf.v, (BB)[t], acc[t]);       \
  __syncthreads(); }

// Fused pass: read scores (NT, streamed once), write 0/1 mask in-place (NT),
// online softmax, P*Vt via WMMA. 2-deep ping-pong: score/V loads for kt+2/kt+3
// are in flight while tile kt/kt+1 is processed.
__global__ __launch_bounds__(128) void attn_out(float* __restrict__ sc,
                                                const float* __restrict__ thr,
                                                const unsigned short* __restrict__ Vt,
                                                float* __restrict__ out) {
  __shared__ float rowmax[4][16], rowsum[4][16], rowscale[4][16];
  int lane = threadIdx.x & 31, wave = threadIdx.x >> 5;
  int m = lane & 15, half = lane >> 4;
  int bh = blockIdx.z, b = bh >> 4, h = bh & 15;
  int qbase = (blockIdx.x * 4 + wave) * 16;
  float th = thr[bh];

  if (half == 0) { rowmax[wave][m] = -3.0e38f; rowsum[wave][m] = 0.0f; rowscale[wave][m] = 1.0f; }
  __syncthreads();

  v8f zero = {0.f,0.f,0.f,0.f,0.f,0.f,0.f,0.f};
  v8f acc[4] = {zero, zero, zero, zero};
  float* srow = sc + (size_t)bh * S_LEN * S_LEN + (size_t)(qbase + m) * S_LEN;
  const unsigned short* Vb = Vt + (size_t)bh * HD * S_LEN;

  float sv0[16], sv1[16];
  v16bf bb0[4], bb1[4];
  LOAD_SCORES(sv0, 0) LOAD_VFRAG(bb0, 0)
  LOAD_SCORES(sv1, 1) LOAD_VFRAG(bb1, 1)

  const int NT = S_LEN / 32;               // 64 tiles
  for (int kt = 0; kt < NT - 2; kt += 2) {
    PROC_TILE(sv0, bb0, kt)
    LOAD_SCORES(sv0, kt + 2) LOAD_VFRAG(bb0, kt + 2)
    PROC_TILE(sv1, bb1, kt + 1)
    LOAD_SCORES(sv1, kt + 3) LOAD_VFRAG(bb1, kt + 3)
  }
  PROC_TILE(sv0, bb0, NT - 2)
  PROC_TILE(sv1, bb1, NT - 1)

  float rs[8];
#pragma unroll
  for (int j = 0; j < 8; ++j) rs[j] = rowsum[wave][j + 8 * half];
#pragma unroll
  for (int t = 0; t < 4; ++t)
#pragma unroll
    for (int j = 0; j < 8; ++j) {
      int q   = qbase + j + 8 * half;
      int col = h * HD + t * 16 + m;
      __builtin_nontemporal_store(acc[t][j] / rs[j],
                                  &out[((size_t)b * S_LEN + q) * EMB + col]);
    }
}

extern "C" void kernel_launch(void* const* d_in, const int* in_sizes, int n_in,
                              void* d_out, int out_size, void* d_ws, size_t ws_size,
                              hipStream_t stream) {
  const float* x  = (const float*)d_in[0];
  const float* Wq = (const float*)d_in[1];
  const float* bq = (const float*)d_in[2];
  const float* Wk = (const float*)d_in[3];
  const float* bk = (const float*)d_in[4];
  const float* Wv = (const float*)d_in[5];
  const float* bv = (const float*)d_in[6];

  float* outp = (float*)d_out;
  float* mask = outp + (size_t)BATCH * S_LEN * EMB;   // S×S map: scores scratch -> mask

  const size_t NX = (size_t)BATCH * S_LEN * EMB;      // 4194304
  const size_t NW = (size_t)EMB * EMB;                // 1048576
  unsigned short* xbf  = (unsigned short*)d_ws;
  unsigned short* wqbf = xbf  + NX;
  unsigned short* wkbf = wqbf + NW;
  unsigned short* wvbf = wkbf + NW;
  unsigned short* qbf  = wvbf + NW;
  unsigned short* kbf  = qbf  + NX;
  unsigned short* vtbf = kbf  + NX;                   // [b,h,d,S]
  int*   hist = (int*)(vtbf + NX);
  float* thr  = (float*)(hist + BHCNT * NBINS);

  cvt_bf16<<<dim3((unsigned)((NX + 255) / 256)), 256, 0, stream>>>(x,  xbf,  (int)NX);
  cvt_bf16<<<dim3((unsigned)((NW + 255) / 256)), 256, 0, stream>>>(Wq, wqbf, (int)NW);
  cvt_bf16<<<dim3((unsigned)((NW + 255) / 256)), 256, 0, stream>>>(Wk, wkbf, (int)NW);
  cvt_bf16<<<dim3((unsigned)((NW + 255) / 256)), 256, 0, stream>>>(Wv, wvbf, (int)NW);
  zero_i32<<<dim3((BHCNT * NBINS + 255) / 256), 256, 0, stream>>>(hist, BHCNT * NBINS);

  // M=4096 rows, N=1024 cols, block tile 32x256 (4 waves x 32x64), K ping-pong
  gemm_qkv<<<dim3(128, 4, 1), 128, 0, stream>>>(xbf, wqbf, bq, qbf, 0);
  gemm_qkv<<<dim3(128, 4, 1), 128, 0, stream>>>(xbf, wkbf, bk, kbf, 0);
  gemm_qkv<<<dim3(128, 4, 1), 128, 0, stream>>>(xbf, wvbf, bv, vtbf, 1);

  // scores: per (b,h), 2048x2048, block tile 16x256
  scores_hist<<<dim3(128, 8, BHCNT), 128, 0, stream>>>(qbf, kbf, mask, hist);
  find_thresh<<<dim3(BHCNT), 32, 0, stream>>>(hist, thr);

  // fused mask + softmax + P*V: 4 waves/block, each 16 q-rows, 2-deep pipeline
  attn_out<<<dim3(32, 1, BHCNT), 128, 0, stream>>>(mask, thr, vtbf, outp);
}
